// NN_91036126806269
// MI455X (gfx1250) — matrix operand order repaired and checked by reading, not compiled, not run on hardware
//
#include <hip/hip_runtime.h>
#include <hip/hip_bf16.h>
#include <math.h>

// ---------------------------------------------------------------------------
// MI455X (gfx1250) implementation, wave32 + v_wmma_f32_16x16x32_f16 + TDM.
// Shapes: B=192, Y=32, Q=12, K=64, D=300 (pad 320), F=2048, V=30000.
// ---------------------------------------------------------------------------

typedef __attribute__((ext_vector_type(16))) _Float16 v16h;
typedef __attribute__((ext_vector_type(8)))  _Float16 v8h;
typedef __attribute__((ext_vector_type(8)))  float    v8f;
typedef __attribute__((ext_vector_type(4)))  float    v4f;
typedef __attribute__((ext_vector_type(4)))  unsigned int u32x4;
typedef __attribute__((ext_vector_type(8)))  int      i32x8;
typedef __attribute__((ext_vector_type(4)))  int      i32x4;

#define DB   192
#define DY   32
#define DQ   12
#define DK   64
#define DD   300
#define DP   320
#define DF   2048

#if __has_builtin(__builtin_amdgcn_tensor_load_to_lds) && __has_builtin(__builtin_amdgcn_s_wait_tensorcnt)
#define HAVE_TDM 1
#else
#define HAVE_TDM 0
#endif

#define WMMA_F16(a, b, c) \
  __builtin_amdgcn_wmma_f32_16x16x32_f16(false, (a), false, (b), (short)0, (c), false, false)

static __device__ __forceinline__ v16h pack16(v8h x, v8h y) {
  v16h r;
#pragma unroll
  for (int i = 0; i < 8; ++i) { r[i] = x[i]; r[i + 8] = y[i]; }
  return r;
}

// A fragment (16x32 f16): lane<16 holds K {0..7,16..23} of row (lane&15),
// lane>=16 holds K {8..15,24..31}.
static __device__ __forceinline__ v16h load_fragA_h(const _Float16* rowp, int dd, int lane) {
  const int ka = (lane < 16) ? 0 : 8;
  v8h x = *(const v8h*)(rowp + dd + ka);
  v8h y = *(const v8h*)(rowp + dd + 16 + ka);
  return pack16(x, y);
}

// B fragment (32x16 f16): lanes 0-15 hold K {0..7,16..23} of column (lane&15), lanes 16-31 K {8..15,24..31}.
static __device__ __forceinline__ v16h load_fragB_h(const _Float16* colp, int dd, int lane) {
  const int kb = (lane < 16) ? 0 : 8;
  v8h x = *(const v8h*)(colp + dd + kb);
  v8h y = *(const v8h*)(colp + dd + 16 + kb);
  return pack16(x, y);
}

static __device__ __forceinline__ v8h cvt8(const float* p) {
  v4f a = *(const v4f*)p;
  v4f b = *(const v4f*)(p + 4);
  v8h r;
#pragma unroll
  for (int i = 0; i < 4; ++i) { r[i] = (_Float16)a[i]; r[i + 4] = (_Float16)b[i]; }
  return r;
}

static __device__ __forceinline__ v16h load_fragA_f32(const float* rowp, int kk, int lane) {
  const int ka = (lane < 16) ? 0 : 8;
  return pack16(cvt8(rowp + kk + ka), cvt8(rowp + kk + 16 + ka));
}

// ---------------------------------------------------------------------------
// TDM: DMA a [rows x 320] f16 tile (contiguous, stride 320) from global to LDS.
// D# built per ISA 8.3/8.4: group0 = {flags, lds_addr, global_addr, type=2},
// group1 = {mask/data_size, dims, tile dims, stride}.  2-D tile => tile_dim2=0,
// groups 2/3 zero.  Issued once per wave (EXEC ignored by TDM).
// ---------------------------------------------------------------------------
static __device__ __forceinline__ void tdm_load_tile(unsigned lds_byte,
                                                     unsigned long long gaddr,
                                                     int rows) {
#if HAVE_TDM
  u32x4 g0;
  g0[0] = 1u;                                             // count=1, user mode
  g0[1] = lds_byte;                                       // LDS byte address
  g0[2] = (unsigned)(gaddr & 0xffffffffull);              // global_addr[31:0]
  g0[3] = (unsigned)((gaddr >> 32) & 0x01ffffffull) | (2u << 30);  // [56:32] | type=2
  i32x8 g1;
  const int td0 = DP, tile0 = DP, stride0 = DP;           // elements (2B units)
  g1[0] = (1 << 16);                                      // wg_mask=0, data_size=2B
  g1[1] = (td0 & 0xffff) << 16;                           // tensor_dim0[15:0]
  g1[2] = ((td0 >> 16) & 0xffff) | ((rows & 0xffff) << 16);   // dim0[31:16] | dim1[15:0]
  g1[3] = ((rows >> 16) & 0xffff) | (tile0 << 16);        // dim1[31:16] | tile_dim0
  g1[4] = (rows & 0xffff);                                // tile_dim1 | tile_dim2=0
  g1[5] = stride0;                                        // tensor_dim0_stride[31:0]
  g1[6] = 0;
  g1[7] = 0;
  i32x4 z4 = {0, 0, 0, 0};
#if __has_include(<hip/amd_detail/amd_gfx1250_TDM.h>)
  i32x8 z8 = {0, 0, 0, 0, 0, 0, 0, 0};
  __builtin_amdgcn_tensor_load_to_lds(g0, g1, z4, z4, z8, 0);   // clang-23 form
#else
  __builtin_amdgcn_tensor_load_to_lds(g0, g1, z4, z4, 0);        // ROCm 7.2 form
#endif
#else
  (void)lds_byte; (void)gaddr; (void)rows;
#endif
}

// ---------------------------------------------------------------------------
// Kernel 0a: Weff = Wp + 1e-5*Wm (f16, zero-padded 320x320), beff = bp + 1e-5*bm
// ---------------------------------------------------------------------------
__global__ void prep_weff(const float* __restrict__ Wp, const float* __restrict__ Wm,
                          const float* __restrict__ bp, const float* __restrict__ bm,
                          _Float16* __restrict__ Weff, float* __restrict__ beff) {
  int idx = blockIdx.x * 256 + threadIdx.x;
  if (idx < DP * DP) {
    int e = idx / DP, d = idx - e * DP;
    float v = (e < DD && d < DD) ? (Wp[e * DD + d] + 1e-5f * Wm[e * DD + d]) : 0.f;
    Weff[idx] = (_Float16)v;
  }
  if (idx < DD) beff[idx] = bp[idx] + 1e-5f * bm[idx];
}

// ---------------------------------------------------------------------------
// Kernel 0b: Wf -> f16, zero-padded to [320][2048] (kills guards + repeated cvt)
// ---------------------------------------------------------------------------
__global__ void prep_wfh(const float* __restrict__ Wf, _Float16* __restrict__ Wfh) {
  int idx = blockIdx.x * 256 + threadIdx.x;
  if (idx < DP * DF) {
    int e = idx >> 11, f = idx & (DF - 1);
    Wfh[idx] = (e < DD) ? (_Float16)Wf[e * DF + f] : (_Float16)0.f;
  }
}

// ---------------------------------------------------------------------------
// Kernel 1: k_emb = wv[label] + feature @ Wf^T + bf   -> f16 [B*K, 320]
// GEMM M=12288 N=320 Kdim=2048.  One wave per 16x80 C strip (5 WMMA tiles):
// the A fragment (f32->f16 cvt) is amortized over 5 WMMAs, B is pre-padded f16.
// ---------------------------------------------------------------------------
__global__ void kemb_gemm(const float* __restrict__ feature, const _Float16* __restrict__ Wfh,
                          const int* __restrict__ label, const float* __restrict__ wv,
                          const float* __restrict__ bf, _Float16* __restrict__ kemb) {
  const int lane = threadIdx.x;
  const int row0 = blockIdx.x * 16;                 // 768 M tiles
  const int col0 = blockIdx.y * 80;                 // 4 N strips of 5 tiles
  const float* arow = feature + (size_t)(row0 + (lane & 15)) * DF;
  const _Float16* bcol[5];
#pragma unroll
  for (int t = 0; t < 5; ++t)
    bcol[t] = Wfh + (size_t)(col0 + t * 16 + (lane & 15)) * DF;

  v8f c[5] = {};
  for (int kk = 0; kk < DF; kk += 32) {
    __builtin_prefetch(arow + kk + 128, 0, 3);
    v16h a = load_fragA_f32(arow, kk, lane);
#pragma unroll
    for (int t = 0; t < 5; ++t) {
      v16h b = load_fragB_h(bcol[t], kk, lane);
      c[t] = WMMA_F16(a, b, c[t]);
    }
  }
#pragma unroll
  for (int t = 0; t < 5; ++t) {
    const int col = col0 + t * 16 + (lane & 15);
    const bool cv = col < DD;
#pragma unroll
    for (int r = 0; r < 8; ++r) {
      int row = row0 + r + ((lane >> 4) << 3);      // C layout: lanes>=16 -> M+8
      float v = 0.f;
      if (cv) {
        int lab = label[row];
        v = c[t][r] + wv[(size_t)lab * DD + col] + bf[col];
      }
      kemb[(size_t)row * DP + col] = (_Float16)v;
    }
  }
}

// ---------------------------------------------------------------------------
// Kernel 2: per (b,y): scores[q,k] via WMMA 16x64x320;
// max_k softmax_k == 1/sum_k exp(s - smax); softmax over q; p_raw = sum_q w_q*wv[...]
// ---------------------------------------------------------------------------
__global__ void att_pemb(const int* __restrict__ query, const float* __restrict__ wv,
                         const _Float16* __restrict__ kemb, _Float16* __restrict__ praw) {
  const int lane = threadIdx.x;
  const int by = blockIdx.x;           // b*32 + y
  const int b  = by >> 5;
  __shared__ float lds_q[12];
  __shared__ float lds_w[12];
  __shared__ int   lds_id[12];
  if (lane < DQ) lds_id[lane] = query[by * DQ + lane];
  __syncthreads();

  const int q   = lane & 15;
  const bool qv = q < DQ;
  const float* qrow = wv + (size_t)(qv ? lds_id[q] : 0) * DD;
  const _Float16* krow[4];
#pragma unroll
  for (int t = 0; t < 4; ++t) krow[t] = kemb + (size_t)(b * DK + t * 16 + (lane & 15)) * DP;

  v8f c[4] = {};
  const int ka = (lane < 16) ? 0 : 8;
  for (int dd = 0; dd < DP; dd += 32) {
    v16h a;
#pragma unroll
    for (int i = 0; i < 8; ++i) {
      int d0 = dd + ka + i, d1 = dd + 16 + ka + i;
      a[i]     = (qv && d0 < DD) ? (_Float16)qrow[d0] : (_Float16)0.f;
      a[i + 8] = (qv && d1 < DD) ? (_Float16)qrow[d1] : (_Float16)0.f;
    }
#pragma unroll
    for (int t = 0; t < 4; ++t) {
      v16h bf16 = load_fragB_h(krow[t], dd, lane);
      c[t] = WMMA_F16(a, bf16, c[t]);
    }
  }

  const float scale = 0.05773502691896258f;  // 1/sqrt(300)
#pragma unroll
  for (int r = 0; r < 8; ++r) {
    int qq = r + ((lane >> 4) << 3);
    float s0 = c[0][r] * scale, s1 = c[1][r] * scale;
    float s2 = c[2][r] * scale, s3 = c[3][r] * scale;
    float m = fmaxf(fmaxf(s0, s1), fmaxf(s2, s3));
    for (int off = 1; off < 16; off <<= 1) m = fmaxf(m, __shfl_xor(m, off));
    float e = __expf(s0 - m) + __expf(s1 - m) + __expf(s2 - m) + __expf(s3 - m);
    for (int off = 1; off < 16; off <<= 1) e += __shfl_xor(e, off);
    if (qq < DQ && (lane & 15) == 0) lds_q[qq] = 1.f / e;   // max of softmax row
  }
  __syncthreads();
  if (lane < DQ) {
    float mx = -1e30f;
    for (int i = 0; i < DQ; ++i) mx = fmaxf(mx, lds_q[i]);
    float sum = 0.f;
    for (int i = 0; i < DQ; ++i) sum += __expf(lds_q[i] - mx);
    lds_w[lane] = __expf(lds_q[lane] - mx) / sum;
  }
  __syncthreads();
  for (int dblk = 0; dblk < 10; ++dblk) {
    int d = dblk * 32 + lane;
    float acc = 0.f;
    if (d < DD) {
#pragma unroll
      for (int i = 0; i < DQ; ++i) acc += lds_w[i] * wv[(size_t)lds_id[i] * DD + d];
    }
    praw[(size_t)by * DP + d] = (_Float16)acc;
  }
}

// ---------------------------------------------------------------------------
// Kernel 3: p_final = p_raw @ Weff^T + beff  -> f16 [B*Y, 320]
// ---------------------------------------------------------------------------
__global__ void plin_gemm(const _Float16* __restrict__ praw, const _Float16* __restrict__ Weff,
                          const float* __restrict__ beff, _Float16* __restrict__ pfin) {
  const int lane = threadIdx.x;
  const int row0 = blockIdx.x * 16;                 // 0..383
  const int col  = blockIdx.y * 16 + (lane & 15);   // 0..19
  const _Float16* arow = praw + (size_t)(row0 + (lane & 15)) * DP;
  const _Float16* bcol = Weff + (size_t)col * DP;   // rows e>=300 are zero
  v8f c = {};
  for (int dd = 0; dd < DP; dd += 32) {
    v16h a = load_fragA_h(arow, dd, lane);
    v16h b = load_fragB_h(bcol, dd, lane);
    c = WMMA_F16(a, b, c);
  }
  const bool cv = col < DD;
  const float bb = cv ? beff[col] : 0.f;
#pragma unroll
  for (int r = 0; r < 8; ++r) {
    int row = row0 + r + ((lane >> 4) << 3);
    float v = cv ? (c[r] + bb) : 0.f;
    pfin[(size_t)row * DP + col] = (_Float16)v;
  }
}

// ---------------------------------------------------------------------------
// Kernel 4 (fused): per (b,a) attmap tile S[32,64] = p_final[b] . k_emb[a]^T
// TDM stages k_emb[a] (40 KB) and p_final[b] (20 KB) into LDS via
// tensor_load_to_lds (one descriptor per wave), s_wait_tensorcnt, then WMMA
// from LDS; softmax over v -> att_obj_sum, rowmax -> logits, all in-register.
// attmap (302 MB) never touches memory.
// ---------------------------------------------------------------------------
__global__ void score_kernel(const _Float16* __restrict__ pfin, const _Float16* __restrict__ kemb,
                             const int* __restrict__ num_query,
                             float* __restrict__ logits, float* __restrict__ target,
                             float* __restrict__ attsum) {
  const int a = blockIdx.x, b = blockIdx.y;
  const int lane = threadIdx.x & 31;
  const int w = threadIdx.x >> 5;                   // 2 waves: rows 0-15 / 16-31
  __shared__ float lds_att[2][64];
  __shared__ float lds_max[32];

#if HAVE_TDM
  __shared__ _Float16 tile[(DK + DY) * DP];         // 60 KB: k-tile then p-tile
  const unsigned lds_base = (unsigned)(uintptr_t)&tile[0];
  if (w == 0) {
    tdm_load_tile(lds_base, (unsigned long long)(uintptr_t)(kemb + (size_t)a * DK * DP), DK);
  } else {
    tdm_load_tile(lds_base + DK * DP * 2,
                  (unsigned long long)(uintptr_t)(pfin + (size_t)b * DY * DP), DY);
  }
  __builtin_amdgcn_s_wait_tensorcnt(0);
  __syncthreads();
  const _Float16* arow = &tile[DK * DP] + (size_t)(w * 16 + (lane & 15)) * DP;
  const _Float16* krow[4];
#pragma unroll
  for (int t = 0; t < 4; ++t) krow[t] = &tile[0] + (size_t)(t * 16 + (lane & 15)) * DP;
#else
  const _Float16* arow = pfin + (size_t)(b * DY + w * 16 + (lane & 15)) * DP;
  const _Float16* krow[4];
#pragma unroll
  for (int t = 0; t < 4; ++t) krow[t] = kemb + (size_t)(a * DK + t * 16 + (lane & 15)) * DP;
#endif

  v8f c[4] = {};
  for (int dd = 0; dd < DP; dd += 32) {
    v16h afrag = load_fragA_h(arow, dd, lane);
#pragma unroll
    for (int t = 0; t < 4; ++t) {
      v16h bfrag = load_fragB_h(krow[t], dd, lane);
      c[t] = WMMA_F16(afrag, bfrag, c[t]);
    }
  }

  float acc[4] = {0.f, 0.f, 0.f, 0.f};
#pragma unroll
  for (int r = 0; r < 8; ++r) {
    float m = fmaxf(fmaxf(c[0][r], c[1][r]), fmaxf(c[2][r], c[3][r]));
    for (int off = 1; off < 16; off <<= 1) m = fmaxf(m, __shfl_xor(m, off));
    float e0 = __expf(c[0][r] - m), e1 = __expf(c[1][r] - m);
    float e2 = __expf(c[2][r] - m), e3 = __expf(c[3][r] - m);
    float es = e0 + e1 + e2 + e3;
    for (int off = 1; off < 16; off <<= 1) es += __shfl_xor(es, off);
    float inv = 1.f / es;
    acc[0] += e0 * inv; acc[1] += e1 * inv; acc[2] += e2 * inv; acc[3] += e3 * inv;
    if ((lane & 15) == 0) lds_max[w * 16 + r + ((lane >> 4) << 3)] = m;
  }
#pragma unroll
  for (int t = 0; t < 4; ++t) {
    acc[t] += __shfl_xor(acc[t], 16);               // combine row halves
    if (lane < 16) lds_att[w][t * 16 + lane] = acc[t];
  }
  __syncthreads();
  if (w == 0) {
#pragma unroll
    for (int i = 0; i < 2; ++i) {
      int v = lane + 32 * i;
      attsum[((size_t)b * DB + a) * DK + v] = lds_att[0][v] + lds_att[1][v];
    }
    float s = lds_max[lane];
    for (int off = 1; off < 32; off <<= 1) s += __shfl_xor(s, off);
    if (lane == 0) {
      logits[b * DB + a] = s / (float)num_query[b];
      target[b * DB + a] = (a == b) ? 1.f : 0.f;
    }
  }
}

// ---------------------------------------------------------------------------
// Host launcher
// ---------------------------------------------------------------------------
static inline size_t align256(size_t x) { return (x + 255) & ~(size_t)255; }

extern "C" void kernel_launch(void* const* d_in, const int* in_sizes, int n_in,
                              void* d_out, int out_size, void* d_ws, size_t ws_size,
                              hipStream_t stream) {
  (void)in_sizes; (void)n_in; (void)out_size; (void)ws_size;
  const int*   query     = (const int*)d_in[0];
  const int*   label     = (const int*)d_in[1];
  const float* feature   = (const float*)d_in[2];
  const int*   num_query = (const int*)d_in[3];
  const float* wv        = (const float*)d_in[4];
  const float* Wp        = (const float*)d_in[5];
  const float* bp        = (const float*)d_in[6];
  const float* Wf        = (const float*)d_in[7];
  const float* bf        = (const float*)d_in[8];
  const float* Wm        = (const float*)d_in[9];
  const float* bm        = (const float*)d_in[10];

  float* out    = (float*)d_out;
  float* logits = out;                         // [192,192]
  float* target = out + DB * DB;               // [192,192]
  float* attsum = out + 2 * DB * DB;           // [192,192,64]

  char* ws = (char*)d_ws;
  size_t off = 0;
  _Float16* kemb = (_Float16*)(ws + off); off += align256((size_t)DB * DK * DP * 2);
  _Float16* praw = (_Float16*)(ws + off); off += align256((size_t)DB * DY * DP * 2);
  _Float16* pfin = (_Float16*)(ws + off); off += align256((size_t)DB * DY * DP * 2);
  _Float16* Weff = (_Float16*)(ws + off); off += align256((size_t)DP * DP * 2);
  _Float16* Wfh  = (_Float16*)(ws + off); off += align256((size_t)DP * DF * 2);
  float*    beff = (float*)(ws + off);    off += align256((size_t)DD * 4);

  prep_weff<<<(DP * DP + 255) / 256, 256, 0, stream>>>(Wp, Wm, bp, bm, Weff, beff);
  prep_wfh<<<(DP * DF + 255) / 256, 256, 0, stream>>>(Wf, Wfh);
  kemb_gemm<<<dim3(DB * DK / 16, 4), 32, 0, stream>>>(feature, Wfh, label, wv, bf, kemb);
  att_pemb<<<DB * DY, 32, 0, stream>>>(query, wv, kemb, praw);
  plin_gemm<<<dim3(DB * DY / 16, DP / 16), 32, 0, stream>>>(praw, Weff, beff, pfin);
  score_kernel<<<dim3(DB, DB), 64, 0, stream>>>(pfin, kemb, num_query, logits, target, attsum);
}
